// SPN_structure_4226247819586
// MI455X (gfx1250) — matrix operand order repaired and checked by reading, not compiled, not run on hardware
//
#include <hip/hip_runtime.h>
#include <hip/hip_bf16.h>
#include <math.h>

typedef _Float16 v16h __attribute__((ext_vector_type(16)));
typedef _Float16 v8h  __attribute__((ext_vector_type(8)));
typedef float    v8f  __attribute__((ext_vector_type(8)));

#define SPN_B     32768
#define SPN_XDIM  32
#define SPN_S     16
#define SPN_SS    256
#define SPN_NC    30   // combines with OC = 16 (block0 + 29 middle)

// ---------------------------------------------------------------------------
// Prep 1: per (var, bit) Bernoulli leaf rows, pre-exponentiated:
//   row[s]   = log_sigmoid(bit ? l : -l)
//   bmax_tab[v*2+bit]       = max_s row[s]
//   ebh_tab[(v*2+bit)*16+s] = (f16) exp(row[s] - rowmax)
// 64 threads = 32 vars * 2 bit states.
// ---------------------------------------------------------------------------
__global__ void spn_prep_leaf(const float* __restrict__ leaf_logits,
                              _Float16* __restrict__ ebh_tab,
                              float* __restrict__ bmax_tab)
{
    int t = threadIdx.x;              // 64
    if (t >= 64) return;
    int v = t >> 1, bit = t & 1;
    float row[SPN_S];
    float mx = -1e30f;
    for (int s = 0; s < SPN_S; ++s) {
        float z   = leaf_logits[v * SPN_S + s];
        float az  = fabsf(z);
        float l1p = __logf(1.0f + __expf(-az));   // log1p(exp(-|z|)), stable
        float lp  = bit ? (fminf(z, 0.0f) - l1p) : (fminf(-z, 0.0f) - l1p);
        row[s] = lp;
        mx = fmaxf(mx, lp);
    }
    bmax_tab[t] = mx;
    for (int s = 0; s < SPN_S; ++s)
        ebh_tab[t * SPN_S + s] = (_Float16)__expf(row[s] - mx);
}

// ---------------------------------------------------------------------------
// Prep 2: per-step column softmax of W[256][16] -> f16, pre-packed into the
// wave32 WMMA B-fragment layout (32x16 f16 B-matrix):
//   lane L holds column n = L&15; lanes 0-15 hold K=0..7,16..23, lanes 16-31 K=8..15,24..31,
//   half h within the fragment <-> K = kbase + (h&7) + 16*(h>>3).
// bfrag layout: [step][k0 chunk 0..7][lane 0..31][h 0..15]  (f16)
// Block 30 normalizes the root weights W_last[256] -> wlast (f32).
// ---------------------------------------------------------------------------
__global__ void spn_prep_weights(const float* __restrict__ sw,
                                 const float* __restrict__ swlast,
                                 _Float16* __restrict__ bfrag,
                                 float* __restrict__ wlast)
{
    __shared__ _Float16 wl[SPN_SS][SPN_S];   // normalized weights, f16
    __shared__ float    red[SPN_SS];

    int t   = blockIdx.x;
    int tid = threadIdx.x;                   // 256

    if (t < SPN_NC) {
        const float* W = sw + (size_t)t * SPN_SS * SPN_S;
        if (tid < SPN_S) {
            int oc = tid;
            float mx = -1e30f;
            for (int k = 0; k < SPN_SS; ++k) mx = fmaxf(mx, W[k * SPN_S + oc]);
            float sum = 0.0f;
            for (int k = 0; k < SPN_SS; ++k) sum += __expf(W[k * SPN_S + oc] - mx);
            float inv = 1.0f / sum;
            for (int k = 0; k < SPN_SS; ++k)
                wl[k][oc] = (_Float16)(__expf(W[k * SPN_S + oc] - mx) * inv);
        }
        __syncthreads();
        // pack: 256 threads = 8 chunks * 32 lanes, 16 halves each
        int lane = tid & 31;
        int k0   = tid >> 5;
        int n    = lane & 15;
        int kbase = 32 * k0 + ((lane >= 16) ? 8 : 0);
        _Float16* dst = bfrag + ((size_t)(t * 8 + k0) * 32 + lane) * 16;
        #pragma unroll
        for (int h = 0; h < 16; ++h)
            dst[h] = wl[kbase + (h & 7) + ((h >> 3) << 4)][n];
    } else {
        // root weights: softmax over 256, keep f32
        float v = swlast[tid];
        red[tid] = v; __syncthreads();
        for (int s = 128; s > 0; s >>= 1) {
            if (tid < s) red[tid] = fmaxf(red[tid], red[tid + s]);
            __syncthreads();
        }
        float mx = red[0]; __syncthreads();
        float e = __expf(v - mx);
        red[tid] = e; __syncthreads();
        for (int s = 128; s > 0; s >>= 1) {
            if (tid < s) red[tid] += red[tid + s];
            __syncthreads();
        }
        wlast[tid] = e / red[0];
    }
}

// ---------------------------------------------------------------------------
// Main: each wave32 owns 16 samples, fully decoupled from the other waves.
// Per combine step:
//   Rank-1: exp(a[i]+b[j]-offsets) = eav[i]*ebs[j].
//   Lane half determines which 8 b-factors it ever needs -> load ebs with a
//   +hi8 pointer offset (no lane selects). All afrag indices lane-uniform.
//   8x V_WMMA_F32_16X16X32_F16 chained over K=256, f32 accumulate.
//   LDS holds raw log(acc); the per-row log-offset cancels in the next
//   renormalization, so it is carried as one scalar per lane (aoff) --
//   no basev broadcast, no shuffles, no barriers.
// ---------------------------------------------------------------------------
__global__ __launch_bounds__(256) void spn_main(
    const int* __restrict__ x,
    const _Float16* __restrict__ ebh_tab,
    const float* __restrict__ bmax_tab,
    const _Float16* __restrict__ bfrag,
    const float* __restrict__ wlast,
    float* __restrict__ out)
{
    __shared__ __align__(16) float curr_lds[8][SPN_S][20];  // padded rows: conflict-free

    int tid  = threadIdx.x;
    int wave = tid >> 5;
    int lane = tid & 31;
    int m    = lane & 15;
    int hi8  = (lane >= 16) ? 8 : 0;
    int sample = (blockIdx.x * 8 + wave) * 16 + m;
    const int* xr = x + (size_t)sample * SPN_XDIM;

    // pack this sample's 32 binary values into one mask (8x b128 loads, once)
    unsigned mask = 0;
    #pragma unroll
    for (int w = 0; w < 8; ++w) {
        int4 q = ((const int4*)xr)[w];
        mask |= (unsigned)(q.x & 1) << (4 * w + 0);
        mask |= (unsigned)(q.y & 1) << (4 * w + 1);
        mask |= (unsigned)(q.z & 1) << (4 * w + 2);
        mask |= (unsigned)(q.w & 1) << (4 * w + 3);
    }

    // step 0: both operands are leaves -> both sides from tables
    v16h eav;          // a-side factors (all 16, lane-uniform indexing)
    v8h  ebs;          // b-side factors: this lane-half's 8 values only
    float aoff, boff;  // running log-domain offsets
    float lrow[SPN_S]; // last log(acc) row (needed for the root step)
    float lmax = 0.0f;
    {
        int b0 = (mask >> 0) & 1, b1 = (mask >> 1) & 1;
        eav  = *(const v16h*)(ebh_tab + (size_t)(0 * 2 + b0) * SPN_S);
        aoff = bmax_tab[0 * 2 + b0];
        ebs  = *(const v8h*)(ebh_tab + (size_t)(1 * 2 + b1) * SPN_S + hi8);
        boff = bmax_tab[1 * 2 + b1];
    }

    for (int c = 0; c < SPN_NC; ++c) {
        v8f acc = {0.f, 0.f, 0.f, 0.f, 0.f, 0.f, 0.f, 0.f};
        const _Float16* bf = bfrag + ((size_t)(c * 8) * 32 + lane) * 16;

        #pragma unroll
        for (int k0 = 0; k0 < 8; ++k0) {
            v16h afrag;
            // lane-uniform indices: i = 2*k0 + (h>>3), j = h&7 (ebs pre-offset by hi8)
            #pragma unroll
            for (int h = 0; h < 16; ++h)
                afrag[h] = eav[2 * k0 + (h >> 3)] * ebs[h & 7];
            v16h bfr = *(const v16h*)(bf + (size_t)k0 * 32 * 16);
            acc = __builtin_amdgcn_wmma_f32_16x16x32_f16(
                false, afrag, false, bfr, (short)0, acc, false, false);
        }

        // D layout: VGPR r <-> row r + 8*hiHalf, col = lane&15. Store raw log(acc).
        #pragma unroll
        for (int r = 0; r < 8; ++r)
            curr_lds[wave][r + hi8][m] = __logf(acc[r]);
        __builtin_amdgcn_wave_barrier();   // same-wave DS ops are in-order; fence compiler

        // transpose readback: lane m takes row m (vectorized, conflict-free)
        {
            const float4* rowp = (const float4*)&curr_lds[wave][m][0];
            float4 q0 = rowp[0], q1 = rowp[1], q2 = rowp[2], q3 = rowp[3];
            lrow[0]=q0.x; lrow[1]=q0.y; lrow[2]=q0.z; lrow[3]=q0.w;
            lrow[4]=q1.x; lrow[5]=q1.y; lrow[6]=q1.z; lrow[7]=q1.w;
            lrow[8]=q2.x; lrow[9]=q2.y; lrow[10]=q2.z; lrow[11]=q2.w;
            lrow[12]=q3.x; lrow[13]=q3.y; lrow[14]=q3.z; lrow[15]=q3.w;
        }
        __builtin_amdgcn_wave_barrier();   // keep next iter's stores after these loads

        lmax = -1e30f;
        #pragma unroll
        for (int i = 0; i < SPN_S; ++i)
            lmax = fmaxf(lmax, lrow[i]);
        #pragma unroll
        for (int i = 0; i < SPN_S; ++i)
            eav[i] = (_Float16)__expf(lrow[i] - lmax);   // the only exps in the loop

        aoff = aoff + boff + lmax;         // offset cancellation: carried as one scalar

        // b-side for next step: vars 2..31, straight from tables
        int v  = c + 2;
        int bb = (mask >> v) & 1;
        ebs  = *(const v8h*)(ebh_tab + (size_t)(v * 2 + bb) * SPN_S + hi8);
        boff = bmax_tab[v * 2 + bb];
    }

    // root combine, OC = 1 (rank-1 factorized, split across half-waves)
    float basev = aoff + boff;
    int bb31 = (mask >> 31) & 1;
    v16h ebf = *(const v16h*)(ebh_tab + (size_t)(31 * 2 + bb31) * SPN_S);

    float ea[SPN_S], eb[SPN_S];
    #pragma unroll
    for (int i = 0; i < SPN_S; ++i) {
        ea[i] = __expf(lrow[i] - lmax);    // offset-free (cancels, as in the loop)
        eb[i] = (float)ebf[i];
    }
    // lanes 0-15 handle i = 0..7, lanes 16-31 handle i = 8..15 (rows duplicated)
    float sum = 0.0f;
    #pragma unroll
    for (int r = 0; r < 8; ++r) {
        int i = r + hi8;
        float rowsum = 0.0f;
        #pragma unroll
        for (int j = 0; j < SPN_S; ++j)
            rowsum = fmaf(eb[j], wlast[i * SPN_S + j], rowsum);
        sum = fmaf(ea[i], rowsum, sum);
    }
    sum += __shfl_xor(sum, 16, 32);
    if (lane < 16)
        out[sample] = basev + __logf(sum);
}

// ---------------------------------------------------------------------------
extern "C" void kernel_launch(void* const* d_in, const int* in_sizes, int n_in,
                              void* d_out, int out_size, void* d_ws, size_t ws_size,
                              hipStream_t stream)
{
    const int*   x   = (const int*)d_in[0];    // [B, 32] int32
    const float* ll  = (const float*)d_in[1];  // [32, 16] f32
    const float* sw  = (const float*)d_in[2];  // [30, 256, 16] f32
    const float* swl = (const float*)d_in[3];  // [256, 1] f32
    float* out = (float*)d_out;                // [B] f32

    // workspace carve-up (~249 KB)
    float*    bmax_tab = (float*)d_ws;                 // 64 f32      @ 0
    float*    wlast    = bmax_tab + 64;                // 256 f32     @ 256B
    _Float16* ebh_tab  = (_Float16*)(wlast + 256);     // 1024 f16    @ 1280B (32B aligned)
    _Float16* bfrag    = ebh_tab + 1024;               // 30*8*32*16  @ 3328B (32B aligned)

    spn_prep_leaf<<<1, 64, 0, stream>>>(ll, ebh_tab, bmax_tab);
    spn_prep_weights<<<SPN_NC + 1, 256, 0, stream>>>(sw, swl, bfrag, wlast);
    spn_main<<<SPN_B / 128, 256, 0, stream>>>(x, ebh_tab, bmax_tab, bfrag, wlast, out);
}
